// GatingNetwork_24618752540914
// MI455X (gfx1250) — matrix operand order repaired and checked by reading, not compiled, not run on hardware
//
#include <hip/hip_runtime.h>
#include <hip/hip_bf16.h>
#include <stdint.h>

// Problem constants (match reference)
#define NTOK   32768
#define INDIM  1024
#define HID    256
#define NEXP   64
#define MTILE  128
#define KC     64

typedef __attribute__((ext_vector_type(16))) __bf16 v16bf;
typedef __attribute__((ext_vector_type(8)))  float  v8f;

union Frag { uint4 u[2]; v16bf v; };

// ---- LDS layout (bytes) ---------------------------------------------------
// buf0: sA0 @0      [128][80] bf16 (20480) ; sB0 @20480  [256][80] bf16 (40960)
// buf1: sA1 @61440  (20480)               ; sB1 @81920  (40960)
//   alias @0: sLog [128][68] f32 (34816)  (buf0 dead when logits written)
// sH  @122880 : [128][264] bf16 (67584)
// sW2 @190464 : [ 64][264] bf16 (33792)
// total 224256 B  (< 320KB WGP LDS; one workgroup per WGP)
#define SA0_OFF   0
#define SB0_OFF   20480
#define SA1_OFF   61440
#define SB1_OFF   81920
#define SLOG_OFF  0
#define SH_OFF    122880
#define SW2_OFF   190464
#define LDS_BYTES 224256
#define SA_STR    80
#define SB_STR    80
#define SH_STR    264
#define SW2_STR   264
#define SLOG_STR  68

__device__ __forceinline__ void cvt4(__bf16* d, float4 f) {
  d[0] = (__bf16)f.x; d[1] = (__bf16)f.y; d[2] = (__bf16)f.z; d[3] = (__bf16)f.w;
}

// Issue all global loads for one K-chunk into registers (deep MLW), plus a
// global_prefetch_b8 of the chunk after next.
__device__ __forceinline__ void load_tiles(float4 (&ra)[8], float4 (&rb)[16],
    const float* __restrict__ x, const float* __restrict__ w1,
    int tok0, int k0, int tid) {
  if (k0 + KC < INDIM) {
    int prow = tid >> 1, phalf = tid & 1;   // one 128B line/thread of next x chunk
    __builtin_prefetch(&x[(size_t)(tok0 + prow) * INDIM + k0 + KC + phalf * 32], 0, 1);
  }
  #pragma unroll
  for (int t = 0; t < 8; ++t) {
    int i = tid + t * 256, row = i >> 4, c4 = i & 15;
    ra[t] = ((const float4*)(x + (size_t)(tok0 + row) * INDIM + k0))[c4];
  }
  #pragma unroll
  for (int t = 0; t < 16; ++t) {
    int i = tid + t * 256, row = i >> 4, c4 = i & 15;
    rb[t] = ((const float4*)(w1 + (size_t)row * INDIM + k0))[c4];
  }
}

__device__ __forceinline__ void store_tiles(const float4 (&ra)[8], const float4 (&rb)[16],
    __bf16* sA, __bf16* sB, int tid) {
  #pragma unroll
  for (int t = 0; t < 8; ++t) {
    int i = tid + t * 256, row = i >> 4, c4 = i & 15;
    cvt4(&sA[row * SA_STR + c4 * 4], ra[t]);
  }
  #pragma unroll
  for (int t = 0; t < 16; ++t) {
    int i = tid + t * 256, row = i >> 4, c4 = i & 15;
    cvt4(&sB[row * SB_STR + c4 * 4], rb[t]);
  }
}

// One K-chunk of stage-1 GEMM. Wave grid 4(M)x2(N): 2 A-frags + 8 B-frags
// feed 16 WMMAs; B-frags are software-pipelined two deep.
__device__ __forceinline__ void compute_chunk(const __bf16* sA, const __bf16* sB,
    v8f (&acc)[2][8], int wave_m, int wave_n, int laneLo, int laneHi) {
  #pragma unroll
  for (int kk = 0; kk < KC; kk += 32) {
    Frag a[2];
    #pragma unroll
    for (int mt = 0; mt < 2; ++mt) {
      const int ar = wave_m * 32 + mt * 16 + laneLo;
      const int ac = kk + laneHi * 8;             // A: K 0-7/16-23 vs 8-15/24-31
      a[mt].u[0] = *(const uint4*)&sA[ar * SA_STR + ac];
      a[mt].u[1] = *(const uint4*)&sA[ar * SA_STR + ac + 16];
    }
    const int bc = kk + laneHi * 8;               // B: K 0-7/16-23 vs 8-15/24-31
    Frag b[2];
    {
      const int br = wave_n * 128 + laneLo;
      b[0].u[0] = *(const uint4*)&sB[br * SB_STR + bc];
      b[0].u[1] = *(const uint4*)&sB[br * SB_STR + bc + 16];
    }
    #pragma unroll
    for (int nt = 0; nt < 8; ++nt) {
      const int cur = nt & 1, nxt = cur ^ 1;
      if (nt < 7) {                               // prefetch next B-frag
        const int br = wave_n * 128 + (nt + 1) * 16 + laneLo;
        b[nxt].u[0] = *(const uint4*)&sB[br * SB_STR + bc];
        b[nxt].u[1] = *(const uint4*)&sB[br * SB_STR + bc + 16];
      }
      acc[0][nt] = __builtin_amdgcn_wmma_f32_16x16x32_bf16(
          false, a[0].v, false, b[cur].v, (short)0, acc[0][nt], false, false);
      acc[1][nt] = __builtin_amdgcn_wmma_f32_16x16x32_bf16(
          false, a[1].v, false, b[cur].v, (short)0, acc[1][nt], false, false);
    }
  }
}

__global__ __launch_bounds__(256, 1)
void moe_gate_fused(const float* __restrict__ x,  const float* __restrict__ w1,
                    const float* __restrict__ b1, const float* __restrict__ w2,
                    const float* __restrict__ b2, float* __restrict__ out)
{
  extern __shared__ char smem[];
  __bf16* sA0  = (__bf16*)(smem + SA0_OFF);
  __bf16* sB0  = (__bf16*)(smem + SB0_OFF);
  __bf16* sA1  = (__bf16*)(smem + SA1_OFF);
  __bf16* sB1  = (__bf16*)(smem + SB1_OFF);
  __bf16* sH   = (__bf16*)(smem + SH_OFF);
  __bf16* sW2  = (__bf16*)(smem + SW2_OFF);
  float*  sLog = (float*) (smem + SLOG_OFF);

  const int tid    = threadIdx.x;
  const int wave   = tid >> 5;
  const int lane   = tid & 31;
  const int laneLo = lane & 15;
  const int laneHi = lane >> 4;
  const int wave_m = wave & 3;     // 4-way M split (32 rows each)
  const int wave_n = wave >> 2;    // 2-way N split (128 cols each; stage2: 32)
  const int tok0   = blockIdx.x * MTILE;

  // Prologue: w2 [64][256] fp32 -> bf16 LDS (L2-resident source)
  for (int i = tid; i < (NEXP * HID) / 4; i += 256) {
    int row = i >> 6, c4 = i & 63;
    float4 f = ((const float4*)w2)[row * (HID / 4) + c4];
    cvt4(&sW2[row * SW2_STR + c4 * 4], f);
  }

  v8f acc[2][8] = {};
  float4 ra[8]; float4 rb[16];

  // ---- Stage 1: h = relu(x @ w1^T + b1), double-buffered K pipeline -----
  load_tiles(ra, rb, x, w1, tok0, 0, tid);
  store_tiles(ra, rb, sA0, sB0, tid);
  __syncthreads();

  for (int c = 0; c < INDIM / KC; c += 2) {        // 16 chunks, unrolled x2
    // even chunk: compute buf0 while loading chunk c+1
    load_tiles(ra, rb, x, w1, tok0, (c + 1) * KC, tid);
    compute_chunk(sA0, sB0, acc, wave_m, wave_n, laneLo, laneHi);
    store_tiles(ra, rb, sA1, sB1, tid);
    __syncthreads();
    // odd chunk: compute buf1 while loading chunk c+2
    if (c + 2 < INDIM / KC)
      load_tiles(ra, rb, x, w1, tok0, (c + 2) * KC, tid);
    compute_chunk(sA1, sB1, acc, wave_m, wave_n, laneLo, laneHi);
    if (c + 2 < INDIM / KC)
      store_tiles(ra, rb, sA0, sB0, tid);
    __syncthreads();
  }

  // Epilogue 1: bias + ReLU -> bf16 h tile in LDS
  #pragma unroll
  for (int mt = 0; mt < 2; ++mt) {
    #pragma unroll
    for (int nt = 0; nt < 8; ++nt) {
      const int n = wave_n * 128 + nt * 16 + laneLo;
      const float bias = b1[n];
      #pragma unroll
      for (int r = 0; r < 8; ++r) {
        const int m = wave_m * 32 + mt * 16 + r + laneHi * 8;
        float v = acc[mt][nt][r] + bias;
        sH[m * SH_STR + n] = (__bf16)(v > 0.f ? v : 0.f);
      }
    }
  }
  __syncthreads();

  // ---- Stage 2: logits = h @ w2^T + b2 (wave grid 4Mx2N) ----------------
  v8f acc2[2][2] = {};
  #pragma unroll
  for (int kk = 0; kk < HID; kk += 32) {
    Frag a[2], b[2];
    #pragma unroll
    for (int mt = 0; mt < 2; ++mt) {
      const int ar = wave_m * 32 + mt * 16 + laneLo;
      const int ac = kk + laneHi * 8;
      a[mt].u[0] = *(const uint4*)&sH[ar * SH_STR + ac];
      a[mt].u[1] = *(const uint4*)&sH[ar * SH_STR + ac + 16];
    }
    #pragma unroll
    for (int nt = 0; nt < 2; ++nt) {
      const int br = wave_n * 32 + nt * 16 + laneLo;
      const int bc = kk + laneHi * 8;
      b[nt].u[0] = *(const uint4*)&sW2[br * SW2_STR + bc];
      b[nt].u[1] = *(const uint4*)&sW2[br * SW2_STR + bc + 16];
    }
    #pragma unroll
    for (int mt = 0; mt < 2; ++mt)
      #pragma unroll
      for (int nt = 0; nt < 2; ++nt)
        acc2[mt][nt] = __builtin_amdgcn_wmma_f32_16x16x32_bf16(
            false, a[mt].v, false, b[nt].v, (short)0, acc2[mt][nt], false, false);
  }

  // logits -> LDS (aliases dead buf0 region)
  #pragma unroll
  for (int mt = 0; mt < 2; ++mt) {
    #pragma unroll
    for (int nt = 0; nt < 2; ++nt) {
      const int e = wave_n * 32 + nt * 16 + laneLo;
      const float bias = b2[e];
      #pragma unroll
      for (int r = 0; r < 8; ++r) {
        const int m = wave_m * 32 + mt * 16 + r + laneHi * 8;
        sLog[m * SLOG_STR + e] = acc2[mt][nt][r] + bias;
      }
    }
  }
  __syncthreads();

  // ---- Top-2 + softmax (one thread per token) ---------------------------
  if (tid < MTILE) {
    const float* row = &sLog[tid * SLOG_STR];
    float best = row[0], second = -3.402823466e38f;
    int bi = 0, si = 0;
    #pragma unroll 8
    for (int e = 1; e < NEXP; ++e) {
      float v = row[e];
      if (v > best)        { second = best; si = bi; best = v; bi = e; }
      else if (v > second) { second = v; si = e; }
    }
    const float d  = __expf(second - best);        // stable: second <= best
    const float g0 = 1.f / (1.f + d);
    const float g1 = 1.f - g0;
    const int token = tok0 + tid;
    out[2 * token + 0] = (float)bi;                // top_k_indices (flattened first)
    out[2 * token + 1] = (float)si;
    out[2 * NTOK + 2 * token + 0] = g0;            // top_k_gates
    out[2 * NTOK + 2 * token + 1] = g1;
  }
}

extern "C" void kernel_launch(void* const* d_in, const int* in_sizes, int n_in,
                              void* d_out, int out_size, void* d_ws, size_t ws_size,
                              hipStream_t stream) {
  (void)in_sizes; (void)n_in; (void)d_ws; (void)ws_size; (void)out_size;
  const float* x  = (const float*)d_in[0];
  const float* w1 = (const float*)d_in[1];
  const float* b1 = (const float*)d_in[2];
  const float* w2 = (const float*)d_in[3];
  const float* b2 = (const float*)d_in[4];
  float* out = (float*)d_out;

  (void)hipFuncSetAttribute((const void*)moe_gate_fused,
                            hipFuncAttributeMaxDynamicSharedMemorySize, LDS_BYTES);

  moe_gate_fused<<<NTOK / MTILE, 256, LDS_BYTES, stream>>>(x, w1, b1, w2, b2, out);
}
